// HungarianMatcher_5068061409367
// MI455X (gfx1250) — matrix operand order repaired and checked by reading, not compiled, not run on hardware
//
#include <hip/hip_runtime.h>
#include <hip/hip_bf16.h>
#include <math.h>

typedef __attribute__((ext_vector_type(16))) _Float16 v16h;
typedef __attribute__((ext_vector_type(8)))  float    v8f;

namespace {
constexpr int   BS = 16, NQ = 550, NC = 91, NT = 48, GN = 11;
constexpr int   GQ    = NQ / GN;      // 50 queries per group
constexpr int   MROWS = BS * NQ;      // 8800
constexpr int   NCOLS = BS * NT;      // 768
constexpr int   KPAD  = 96;           // 91 classes padded to 96 (3 x K=32 WMMA steps)
constexpr int   NPAIR = GN * NT;      // 528 assignments per batch
constexpr float WCLS = 2.0f, W3D = 10.0f, WBB = 5.0f, WGI = 2.0f;
constexpr float ALPHA = 0.25f, EPSF = 1e-8f;
constexpr float FINF = 1e30f;
}

// ---------------------------------------------------------------------------
// Kernel 1: focal-loss class-cost table, f32 -> f16, padded K 91->96
//   A[i][k] = pos - neg  (i over bs*nq rows, k over classes)
// ---------------------------------------------------------------------------
__global__ void hm_build_a(const float* __restrict__ logits,
                           _Float16* __restrict__ A) {
  int idx = blockIdx.x * blockDim.x + threadIdx.x;
  if (idx >= MROWS * KPAD) return;
  int i = idx / KPAD, k = idx - i * KPAD;
  float val = 0.0f;
  if (k < NC) {
    float x  = logits[(size_t)i * NC + k];
    float pr = 1.0f / (1.0f + __expf(-x));
    float neg = (1.0f - ALPHA) * pr * pr * (-__logf(1.0f - pr + EPSF));
    float pos = ALPHA * (1.0f - pr) * (1.0f - pr) * (-__logf(pr + EPSF));
    val = pos - neg;
  }
  A[(size_t)i * KPAD + k] = (_Float16)val;
}

// ---------------------------------------------------------------------------
// Kernel 2: one-hot selector matrix, stored transposed: Bt[n][k] = (label[n]==k)
// ---------------------------------------------------------------------------
__global__ void hm_build_b(const int* __restrict__ labels,
                           _Float16* __restrict__ Bt) {
  int idx = blockIdx.x * blockDim.x + threadIdx.x;
  if (idx >= NCOLS * KPAD) return;
  int n = idx / KPAD, k = idx - n * KPAD;
  float v = (k < NC && labels[n] == k) ? 1.0f : 0.0f;
  Bt[(size_t)n * KPAD + k] = (_Float16)v;
}

// ---------------------------------------------------------------------------
// Kernel 3 (fused): cost_class GEMM via v_wmma_f32_16x16x32_f16, with the
// L1-center, L1-bbox and -GIoU terms folded into the epilogue so the 27 MB
// cost matrix is written exactly once. One wave per 16x16 (query x target)
// tile; the 16 pred + 16 tgt boxes for the tile are staged in LDS.
// ---------------------------------------------------------------------------
__global__ void __launch_bounds__(32)
hm_wmma_fused(const _Float16* __restrict__ A, const _Float16* __restrict__ Bt,
              const float* __restrict__ pb_all, const float* __restrict__ tb_all,
              float* __restrict__ out) {
  const int lane = threadIdx.x;
  const int half = lane >> 4;
  const int lo   = lane & 15;
  const int mt = blockIdx.y, ntile = blockIdx.x;

  __shared__ float spb[16][6];   // pred boxes for the tile's 16 rows
  __shared__ float stb[16][6];   // tgt boxes for the tile's 16 cols
  for (int idx = lane; idx < 96; idx += 32) {
    spb[idx / 6][idx % 6] = pb_all[(size_t)(mt * 16) * 6 + idx];
    stb[idx / 6][idx % 6] = tb_all[(size_t)(ntile * 16) * 6 + idx];
  }

  const _Float16* arow = A  + (size_t)(mt * 16 + lo) * KPAD;     // M row
  const _Float16* brow = Bt + (size_t)(ntile * 16 + lo) * KPAD;  // N column
  v8f c = {};
#pragma unroll
  for (int s = 0; s < 3; ++s) {
    const int kb = s * 32;
    union { v16h v; unsigned u[8]; } av, bv;
#pragma unroll
    for (int r = 0; r < 8; ++r) {
      // A 16x32 f16: VGPR r holds K = (r<4?0:16) + half*8 + (r%4)*2, +1
      const int ka = kb + ((r < 4) ? 0 : 16) + half * 8 + (r & 3) * 2;
      av.u[r] = *(const unsigned*)(arow + ka);
      // B 32x16 f16: VGPR r holds K = (r<4?0:16) + half*8 + (r%4)*2, +1
      const int kbq = kb + ((r < 4) ? 0 : 16) + half * 8 + (r & 3) * 2;
      bv.u[r] = *(const unsigned*)(brow + kbq);
    }
    c = __builtin_amdgcn_wmma_f32_16x16x32_f16(false, av.v, false, bv.v,
                                               (short)0, c, false, false);
  }

  __syncthreads();

  // This lane's target box (column lo of the tile), cxcylrtb -> xyxy.
  const float tcx = stb[lo][0], tcy = stb[lo][1];
  const float tl  = stb[lo][2], tr  = stb[lo][3];
  const float tt  = stb[lo][4], tb  = stb[lo][5];
  const float bx0 = tcx - tl, by0 = tcy - tt;
  const float bx1 = tcx + tr, by1 = tcy + tb;
  const float area2 = (bx1 - bx0) * (by1 - by0);

  // D layout: VGPR j -> M = half*8 + j, N = lane%16
  const int col   = ntile * 16 + lo;
  const int rbase = mt * 16 + half * 8;
#pragma unroll
  for (int j = 0; j < 8; ++j) {
    const float* pb = spb[half * 8 + j];
    const float pcx = pb[0], pcy = pb[1];
    const float pl  = pb[2], pr  = pb[3];
    const float pt  = pb[4], pbo = pb[5];

    float c3 = fabsf(pcx - tcx) + fabsf(pcy - tcy);
    float cb = fabsf(pl - tl) + fabsf(pr - tr) +
               fabsf(pt - tt) + fabsf(pbo - tb);

    float ax0 = pcx - pl, ay0 = pcy - pt;
    float ax1 = pcx + pr, ay1 = pcy + pbo;
    float area1 = (ax1 - ax0) * (ay1 - ay0);
    float iw = fminf(ax1, bx1) - fmaxf(ax0, bx0);
    float ih = fminf(ay1, by1) - fmaxf(ay0, by0);
    float inter = fmaxf(iw, 0.0f) * fmaxf(ih, 0.0f);
    float uni = area1 + area2 - inter;
    float iou = inter / uni;
    float ew = fmaxf(fmaxf(ax1, bx1) - fminf(ax0, bx0), 0.0f);
    float eh = fmaxf(fmaxf(ay1, by1) - fminf(ay0, by0), 0.0f);
    float ae = ew * eh;
    float giou = iou - (ae - uni) / ae;

    out[(size_t)(rbase + j) * NCOLS + col] =
        WCLS * c[j] + WBB * cb + W3D * c3 + WGI * (-giou);
  }
}

// ---------------------------------------------------------------------------
// Kernel 4: Jonker-Volgenant LSA, one workgroup per (batch, group).
// The 50x48 block is > square, so (matching reference) solve transposed 48x50:
// rows = targets (n=48), cols = queries (m=50). LDS-resident.
// ---------------------------------------------------------------------------
__global__ void __launch_bounds__(64)
hm_lsa(const float* __restrict__ C, float* __restrict__ out) {
  const int b = blockIdx.y, g = blockIdx.x, tid = threadIdx.x;
  constexpr int N = NT;   // 48 rows (targets)
  constexpr int M = GQ;   // 50 cols (queries)

  __shared__ float cst[N][M];
  __shared__ float su[N + 1], sv[M + 1], sminv[M + 1];
  __shared__ int   sp[M + 1], sway[M + 1], sused[M + 1];
  __shared__ int   s_j0, s_j1, s_done;
  __shared__ float s_delta;

  // Load the transposed cost block into LDS.
  for (int idx = tid; idx < N * M; idx += blockDim.x) {
    int r = idx / M, c = idx - r * M;  // r = target, c = query-in-group
    cst[r][c] = C[((size_t)b * NQ + g * GQ + c) * NCOLS + (size_t)b * NT + r];
  }
  for (int idx = tid; idx <= M; idx += blockDim.x) {
    sv[idx] = 0.0f; sp[idx] = 0; sway[idx] = 0;
    if (idx <= N) su[idx] = 0.0f;
  }
  __syncthreads();

  for (int i = 1; i <= N; ++i) {
    if (tid == 0) { sp[0] = i; s_j0 = 0; s_done = 0; }
    for (int j = tid; j <= M; j += blockDim.x) { sused[j] = 0; sminv[j] = FINF; }
    __syncthreads();
    while (true) {
      if (tid == 0) sused[s_j0] = 1;
      __syncthreads();
      const int   i0  = sp[s_j0];
      const float ui0 = su[i0];
      for (int j = 1 + tid; j <= M; j += blockDim.x) {
        if (!sused[j]) {
          float cur = cst[i0 - 1][j - 1] - ui0 - sv[j];
          if (cur < sminv[j]) { sminv[j] = cur; sway[j] = s_j0; }
        }
      }
      __syncthreads();
      if (tid == 0) {
        float d = FINF; int j1 = 1;
        for (int j = 1; j <= M; ++j)
          if (!sused[j] && sminv[j] < d) { d = sminv[j]; j1 = j; }
        s_j1 = j1; s_delta = d;
      }
      __syncthreads();
      const float delta = s_delta;
      for (int j = tid; j <= M; j += blockDim.x) {
        if (sused[j]) { su[sp[j]] += delta; sv[j] -= delta; }
        else          { sminv[j] -= delta; }
      }
      __syncthreads();
      if (tid == 0) { s_j0 = s_j1; if (sp[s_j0] == 0) s_done = 1; }
      __syncthreads();
      if (s_done) break;
    }
    if (tid == 0) {  // augment
      int j0 = s_j0;
      while (j0 != 0) { int j1 = sway[j0]; sp[j0] = sp[j1]; j0 = j1; }
    }
    __syncthreads();
  }

  // Emit pairs sorted by query index (ascending column == argsort(rows)).
  if (tid == 0) {
    const size_t piBase = (size_t)MROWS * NCOLS;
    const size_t tiBase = piBase + (size_t)BS * NPAIR;
    int cnt = 0;
    for (int c = 1; c <= M; ++c) {
      if (sp[c] != 0) {
        int pos = b * NPAIR + g * NT + cnt;
        out[piBase + pos] = (float)(g * GQ + c - 1);  // pred (row) index
        out[tiBase + pos] = (float)(sp[c] - 1);        // tgt (col) index
        ++cnt;
      }
    }
  }
}

// ---------------------------------------------------------------------------
extern "C" void kernel_launch(void* const* d_in, const int* in_sizes, int n_in,
                              void* d_out, int out_size, void* d_ws, size_t ws_size,
                              hipStream_t stream) {
  const float* logits = (const float*)d_in[0];   // (16,550,91)
  const float* pboxes = (const float*)d_in[1];   // (16,550,6)
  const int*   labels = (const int*)d_in[2];     // (16,48)
  const float* tboxes = (const float*)d_in[3];   // (16,48,6)
  // d_in[4] = group_num (fixed 11 by harness setup; cannot sync-read in capture)
  float* out = (float*)d_out;

  _Float16* A  = (_Float16*)d_ws;                     // 8800 x 96 f16
  _Float16* Bt = A + (size_t)MROWS * KPAD;            // 768  x 96 f16

  hm_build_a<<<(MROWS * KPAD + 255) / 256, 256, 0, stream>>>(logits, A);
  hm_build_b<<<(NCOLS * KPAD + 255) / 256, 256, 0, stream>>>(labels, Bt);

  dim3 gw(NCOLS / 16, MROWS / 16);                    // 48 x 550 tiles
  hm_wmma_fused<<<gw, 32, 0, stream>>>(A, Bt, pboxes, tboxes, out);

  dim3 gl(GN, BS);                                    // 11 x 16 LSA problems
  hm_lsa<<<gl, 64, 0, stream>>>(out, out);
}
